// IOUfitModule_75831942578675
// MI455X (gfx1250) — matrix operand order repaired and checked by reading, not compiled, not run on hardware
//
#include <hip/hip_runtime.h>
#include <hip/hip_bf16.h>

typedef __attribute__((ext_vector_type(16))) _Float16 v16h;
typedef __attribute__((ext_vector_type(8)))  float    v8f;

#define INTER 256

// ---------------------------------------------------------------------------
// Kernel 1: batch-independent precompute.
//   A  = wq.wk, Bk = bq.wk                       (attention scalars)
//   p  = W1 @ (Wo @ wv)/2
//   r  = W1 @ ((Wo @ bv)/2 + bo) + b1
//   W2 repacked to f16 in WMMA B-fragment order (32x16 tiles, contiguous k order chosen here, not the ISA table:
//   lanes 0-15 -> K = 2v+h, lanes 16-31 -> K = 16+2v+h, N = lane%16).
// ---------------------------------------------------------------------------
__global__ __launch_bounds__(256) void iou_precompute(
    const float* __restrict__ Wq, const float* __restrict__ bq,
    const float* __restrict__ Wk, const float* __restrict__ bk,
    const float* __restrict__ Wv, const float* __restrict__ bv,
    const float* __restrict__ Wo, const float* __restrict__ bo,
    const float* __restrict__ W1, const float* __restrict__ b1,
    const float* __restrict__ W2,
    float* __restrict__ ws_f, _Float16* __restrict__ w2h)
{
    __shared__ float red[256];
    __shared__ float wvS[INTER], bvS[INTER], uS[INTER], cS[INTER];
    const int t = threadIdx.x;

    wvS[t] = Wv[t];
    bvS[t] = bv[t];

    const float wk_t = Wk[t];
    red[t] = Wq[t] * wk_t;
    __syncthreads();
    for (int s = 128; s > 0; s >>= 1) { if (t < s) red[t] += red[t + s]; __syncthreads(); }
    const float Aval = red[0];
    __syncthreads();
    red[t] = bq[t] * wk_t;
    __syncthreads();
    for (int s = 128; s > 0; s >>= 1) { if (t < s) red[t] += red[t + s]; __syncthreads(); }
    const float Bkval = red[0];

    // u = 0.5 * Wo @ wv ; c = 0.5 * Wo @ bv + bo
    float u = 0.f, c = 0.f;
    for (int j = 0; j < INTER; ++j) {
        const float w = Wo[t * INTER + j];
        u += w * wvS[j];
        c += w * bvS[j];
    }
    u *= 0.5f;
    c = 0.5f * c + bo[t];
    __syncthreads();
    uS[t] = u; cS[t] = c;
    __syncthreads();

    // p = W1 @ u ; r = W1 @ c + b1
    float p = 0.f, r = 0.f;
    for (int j = 0; j < INTER; ++j) {
        const float w = W1[t * INTER + j];
        p += w * uS[j];
        r += w * cS[j];
    }
    r += b1[t];
    ws_f[t]         = p;
    ws_f[INTER + t] = r;
    if (t == 0) { ws_f[2 * INTER] = Aval; ws_f[2 * INTER + 1] = Bkval; }

    // Repack W2 -> f16, WMMA-B fragment order, per-lane contiguous (16 halfs):
    // idx = ((nt*8 + kt)*32 + lane)*16 + (2v + h)
    for (int idx = t; idx < INTER * INTER; idx += 256) {
        const int e  = idx & 15;
        const int l  = (idx >> 4) & 31;
        const int f  = idx >> 9;     // 0..127
        const int kt = f & 7;
        const int nt = f >> 3;
        const int v  = e >> 1, h = e & 1;
        const int K  = kt * 32 + ((l < 16) ? (2 * v + h) : (16 + 2 * v + h));
        const int N  = nt * 16 + (l & 15);
        w2h[idx] = (_Float16)W2[N * INTER + K];
    }
}

// ---------------------------------------------------------------------------
// Kernel 2: main GEMM. 256 threads = 8 waves; wave handles TWO 16-row M-tiles
// (32 rows = 16 batches), so each B fragment read from LDS feeds two WMMAs.
// Block covers 256 rows = 128 batches.
// ---------------------------------------------------------------------------
__global__ __launch_bounds__(256) void iou_main(
    const float* __restrict__ pred, const float* __restrict__ gt,
    const float* __restrict__ b2,  const float* __restrict__ wf1,
    const float* __restrict__ bf1, const float* __restrict__ wf2,
    const float* __restrict__ bf2,
    const float* __restrict__ ws_f, const _Float16* __restrict__ w2h,
    float* __restrict__ out, int Bn)
{
    __shared__ alignas(32) _Float16 w2s[INTER * INTER];   // 128 KB
    __shared__ float pS[INTER], rS[INTER], b2S[INTER], wfS[INTER];
    __shared__ float aS[256];
    const int t = threadIdx.x;

    // Stage packed W2 (128 KB) into LDS with 16B vector copies.
    {
        const uint4* src = (const uint4*)w2h;
        uint4*       dst = (uint4*)w2s;
        for (int i = t; i < (INTER * INTER * 2) / 16; i += 256) dst[i] = src[i];
    }
    pS[t]  = ws_f[t];
    rS[t]  = ws_f[INTER + t];
    b2S[t] = b2[t];
    wfS[t] = wf1[t];
    const float Aval  = ws_f[2 * INTER];
    const float Bkval = ws_f[2 * INTER + 1];

    const int batchBase = blockIdx.x * 128;

    // Per-row attention scalar a_j (row j = 2*batchLocal + token), 256 rows.
    {
        int bidx = batchBase + (t >> 1);
        if (bidx >= Bn) bidx = Bn - 1;
        const int   s  = t & 1;
        const float x0 = pred[bidx], x1 = gt[bidx];
        const float xs = s ? x1 : x0;
        const float d  = (x1 - x0) * (Aval * xs + Bkval) * (1.0f / 256.0f);
        const float att0 = 1.0f / (1.0f + __expf(d));   // softmax weight on token 0
        aS[t] = x1 + att0 * (x0 - x1);
    }
    __syncthreads();

    const int  wave    = t >> 5, lane = t & 31;
    const int  rowBase = wave * 32;
    const int  m       = lane & 15;
    const bool hiHalf  = lane >= 16;
    const float a0 = aS[rowBase + m];          // M-tile 0: rows rowBase+0..15
    const float a1 = aS[rowBase + 16 + m];     // M-tile 1: rows rowBase+16..31

    // Build 8 A-fragments per M-tile (16x32 f16 each): lanes<16 hold
    // K {0..15}, lanes>=16 hold K {16..31} of each 32-wide block.
    v16h afragA[8], afragB[8];
#pragma unroll
    for (int kt = 0; kt < 8; ++kt) {
        const int base = kt * 32 + (hiHalf ? 16 : 0);
        v16h fA, fB;
#pragma unroll
        for (int e = 0; e < 16; ++e) {
            const int v = e >> 1, h = e & 1;
            const int K = base + (2 * v + h);
            const float pk = pS[K], rk = rS[K];
            float vA = a0 * pk + rk;  vA = vA > 0.f ? vA : 0.f;   // relu(h1)
            float vB = a1 * pk + rk;  vB = vB > 0.f ? vB : 0.f;
            fA[e] = (_Float16)vA;
            fB[e] = (_Float16)vB;
        }
        afragA[kt] = fA;
        afragB[kt] = fB;
    }

    float rowAcc0[8] = {0.f, 0.f, 0.f, 0.f, 0.f, 0.f, 0.f, 0.f};
    float rowAcc1[8] = {0.f, 0.f, 0.f, 0.f, 0.f, 0.f, 0.f, 0.f};
    for (int nt = 0; nt < 16; ++nt) {
        v8f c0 = {}, c1 = {};
#pragma unroll
        for (int kt = 0; kt < 8; ++kt) {
            const v16h bfrag = *(const v16h*)(&w2s[((nt * 8 + kt) * 32 + lane) * 16]);
            c0 = __builtin_amdgcn_wmma_f32_16x16x32_f16(
                false, afragA[kt], false, bfrag, (short)0, c0, false, false);
            c1 = __builtin_amdgcn_wmma_f32_16x16x32_f16(
                false, afragB[kt], false, bfrag, (short)0, c1, false, false);
        }
        // C layout: lane holds N = nt*16 + m; VGPR v -> row v (+8 for hi half).
        const int   N   = nt * 16 + m;
        const float b2n = b2S[N], wn = wfS[N];
#pragma unroll
        for (int v = 0; v < 8; ++v) {
            float x0v = c0[v] + b2n;  x0v = x0v > 0.f ? x0v : 0.f;  // relu(h2)
            float x1v = c1[v] + b2n;  x1v = x1v > 0.f ? x1v : 0.f;
            rowAcc0[v] += x0v * wn;                                 // dot wf1
            rowAcc1[v] += x1v * wn;
        }
    }

    // Reduce the wf1-dot across the 16 lanes of each half-wave.
#pragma unroll
    for (int v = 0; v < 8; ++v) {
        float x = rowAcc0[v];
        x += __shfl_xor(x, 1, 32);
        x += __shfl_xor(x, 2, 32);
        x += __shfl_xor(x, 4, 32);
        x += __shfl_xor(x, 8, 32);
        rowAcc0[v] = x;
        float y = rowAcc1[v];
        y += __shfl_xor(y, 1, 32);
        y += __shfl_xor(y, 2, 32);
        y += __shfl_xor(y, 4, 32);
        y += __shfl_xor(y, 8, 32);
        rowAcc1[v] = y;
    }

    if (m == 0) {   // lanes 0 and 16 hold rows {0..7} and {8..15} of each tile
        const float bf1v = bf1[0], w20 = wf2[0], w21 = wf2[1], bf2v = bf2[0];
        const int mBase = hiHalf ? 8 : 0;
#pragma unroll
        for (int pr = 0; pr < 4; ++pr) {
            // M-tile 0
            {
                const float y0  = rowAcc0[2 * pr]     + bf1v;   // token 0
                const float y1  = rowAcc0[2 * pr + 1] + bf1v;   // token 1
                const int   row = rowBase + mBase + 2 * pr;
                const int bidx  = batchBase + (row >> 1);
                if (bidx < Bn) out[bidx] = y0 * w20 + y1 * w21 + bf2v;
            }
            // M-tile 1
            {
                const float y0  = rowAcc1[2 * pr]     + bf1v;
                const float y1  = rowAcc1[2 * pr + 1] + bf1v;
                const int   row = rowBase + 16 + mBase + 2 * pr;
                const int bidx  = batchBase + (row >> 1);
                if (bidx < Bn) out[bidx] = y0 * w20 + y1 * w21 + bf2v;
            }
        }
    }
}

extern "C" void kernel_launch(void* const* d_in, const int* in_sizes, int n_in,
                              void* d_out, int out_size, void* d_ws, size_t ws_size,
                              hipStream_t stream) {
    const float* pred = (const float*)d_in[0];
    const float* gt   = (const float*)d_in[1];
    const float* Wq   = (const float*)d_in[2];
    const float* bq   = (const float*)d_in[3];
    const float* Wk   = (const float*)d_in[4];
    const float* bk   = (const float*)d_in[5];
    const float* Wv   = (const float*)d_in[6];
    const float* bv   = (const float*)d_in[7];
    const float* Wo   = (const float*)d_in[8];
    const float* bo   = (const float*)d_in[9];
    const float* W1   = (const float*)d_in[10];
    const float* b1   = (const float*)d_in[11];
    const float* W2   = (const float*)d_in[12];
    const float* b2   = (const float*)d_in[13];
    const float* Wf1  = (const float*)d_in[14];
    const float* bf1  = (const float*)d_in[15];
    const float* Wf2  = (const float*)d_in[16];
    const float* bf2  = (const float*)d_in[17];

    float*    out  = (float*)d_out;
    float*    ws_f = (float*)d_ws;                       // p[256], r[256], scalars
    _Float16* w2h  = (_Float16*)((char*)d_ws + 4096);    // 128 KB packed W2 f16

    const int Bn = in_sizes[0];

    iou_precompute<<<1, 256, 0, stream>>>(Wq, bq, Wk, bk, Wv, bv, Wo, bo,
                                          W1, b1, W2, ws_f, w2h);

    const int nblocks = (Bn + 127) / 128;
    iou_main<<<nblocks, 256, 0, stream>>>(pred, gt, b2, Wf1, bf1, Wf2, bf2,
                                          ws_f, w2h, out, Bn);
}
